// SharedTaskMLP_70162585747915
// MI455X (gfx1250) — matrix operand order repaired and checked by reading, not compiled, not run on hardware
//
#include <hip/hip_runtime.h>

typedef __attribute__((ext_vector_type(16))) __bf16 v16bf;
typedef __attribute__((ext_vector_type(8)))  __bf16 v8bf;
typedef __attribute__((ext_vector_type(8)))  float  v8f;
typedef __attribute__((ext_vector_type(4)))  float  v4f;

#define B_TOT 65536
#define DIM   1024
#define TE    16
#define D0    (DIM + TE)   // 1040
#define H1    256
#define H2    64
#define NKT   33           // 32 x-tiles + 1 task-emb tail tile

#define WAVES   8
#define BLOCK_T (WAVES * 32)   // 256 threads
#define M_WG    (WAVES * 16)   // 128 rows per workgroup

// ws (scratch) layout, bf16 elements:
//   wsW1: [33 tiles][256 n][32 k]  (col-major per tile, zero-padded tail)
//   wsW2: [64 n][256 k]            (col-major)
#define WS_W1_ELEMS (NKT * H1 * 32)          // 270336
#define WS_W2_OFF   WS_W1_ELEMS
#define WS_TOTAL    (WS_W1_ELEMS + H1 * H2)  // 286720 elems = 573440 B

// Padded LDS strides (elements) for conflict-free 16B lane accesses.
#define W1T_KS  40     // w1 tile: [256 cols][40 k]     bf16
#define W2T_KS  264    // w2:      [64 cols][264 k]     bf16
#define H1_RS   264    // h1:      [16 rows][264 cols]  bf16, per wave

#define OFF_W2T  0
#define SZ_W2T   (H2 * W2T_KS * 2)          // 33792
#define OFF_W1T0 (OFF_W2T + SZ_W2T)
#define SZ_W1T   (H1 * W1T_KS * 2)          // 20480
#define OFF_W1T1 (OFF_W1T0 + SZ_W1T)
#define OFF_H1   (OFF_W1T1 + SZ_W1T)
#define SZ_H1    (WAVES * 16 * H1_RS * 2)   // 67584
#define SMEM_BYTES (OFF_H1 + SZ_H1)         // 142336 B (< 160KB -> 2 WG / WGP)

__device__ __forceinline__ v16bf cat8(v8bf lo, v8bf hi8) {
  return __builtin_shufflevector(lo, hi8, 0, 1, 2, 3, 4, 5, 6, 7,
                                 8, 9, 10, 11, 12, 13, 14, 15);
}

__device__ __forceinline__ void cvt_f4(v16bf& a, int base, v4f p) {
  a[base + 0] = (__bf16)p[0];
  a[base + 1] = (__bf16)p[1];
  a[base + 2] = (__bf16)p[2];
  a[base + 3] = (__bf16)p[3];
}

// ---------------- one-time weight convert: fp32 -> bf16, tiled col-major ----
__global__ void __launch_bounds__(256)
SharedTaskMLP_wconv_kernel(const float* __restrict__ W1,
                           const float* __restrict__ W2,
                           __bf16* __restrict__ ws) {
  int idx = blockIdx.x * 256 + threadIdx.x;
  if (idx < WS_W1_ELEMS) {
    int t = idx >> 13;      // tile 0..32
    int r = idx & 8191;
    int n = r >> 5;         // 0..255
    int k = r & 31;         // 0..31
    int gk = t * 32 + k;
    float v = (gk < D0) ? W1[gk * H1 + n] : 0.0f;
    ws[idx] = (__bf16)v;
  } else if (idx < WS_TOTAL) {
    int r = idx - WS_W1_ELEMS;  // 0..16383
    int n = r >> 8;             // 0..63
    int k = r & 255;            // 0..255
    ws[idx] = (__bf16)W2[k * H2 + n];
  }
}

// ---------------------------------- main ----------------------------------
__global__ void __launch_bounds__(BLOCK_T)
SharedTaskMLP_70162585747915_kernel(const float* __restrict__ x,
                                    const int* __restrict__ task_ids,
                                    const float* __restrict__ task_emb,
                                    const float* __restrict__ b1,
                                    const float* __restrict__ b2,
                                    const float* __restrict__ head_W,
                                    const float* __restrict__ head_b,
                                    const __bf16* __restrict__ ws,
                                    float* __restrict__ out) {
  extern __shared__ __align__(16) char smem[];
  __bf16* w2T  = (__bf16*)(smem + OFF_W2T);
  __bf16* w1T0 = (__bf16*)(smem + OFF_W1T0);
  __bf16* w1T1 = (__bf16*)(smem + OFF_W1T1);

  const int tid  = threadIdx.x;
  const int wave = tid >> 5;
  const int lane = tid & 31;
  const int l16  = lane & 15;
  const int hi   = lane >> 4;   // half-wave: selects K-half of A/B fragments

  __bf16* h1w = (__bf16*)(smem + OFF_H1) + wave * 16 * H1_RS;

  const int row_wave = blockIdx.x * M_WG + wave * 16;
  const int myrow    = row_wave + l16;
  const float* xrow  = x + (size_t)myrow * DIM;
  const __bf16* wsW2 = ws + WS_W2_OFF;

  // ---- stage W2 into LDS once (bf16 copy, padded stride) ----
  {
    int n = tid >> 2;             // 0..63
    int kc = (tid & 3) * 64;      // 0,64,128,192
    const __bf16* s = wsW2 + n * H1 + kc;
    __bf16* d = w2T + n * W2T_KS + kc;
#pragma unroll
    for (int j = 0; j < 64; j += 8) *(v8bf*)(d + j) = *(const v8bf*)(s + j);
  }

  // ---- layer 1 accumulators ----
  v8f acc1[16];
#pragma unroll
  for (int i = 0; i < 16; ++i)
#pragma unroll
    for (int j = 0; j < 8; ++j) acc1[i][j] = 0.0f;

  const int mytask = task_ids[myrow];

  // ---- prologue: stage tile 0, build A fragment 0 ----
  {
    const __bf16* src = ws + tid * 32;   // tile 0
    v8bf s0 = *(const v8bf*)(src + 0);
    v8bf s1 = *(const v8bf*)(src + 8);
    v8bf s2 = *(const v8bf*)(src + 16);
    v8bf s3 = *(const v8bf*)(src + 24);
    __bf16* d = w1T0 + tid * W1T_KS;
    *(v8bf*)(d + 0)  = s0;
    *(v8bf*)(d + 8)  = s1;
    *(v8bf*)(d + 16) = s2;
    *(v8bf*)(d + 24) = s3;
  }
  v16bf a_cur;
  {
    const float* p = xrow + 8 * hi;
    cvt_f4(a_cur, 0, *(const v4f*)(p + 0));
    cvt_f4(a_cur, 4, *(const v4f*)(p + 4));
    cvt_f4(a_cur, 8, *(const v4f*)(p + 16));
    cvt_f4(a_cur, 12, *(const v4f*)(p + 20));
  }
  __syncthreads();

  // ---- software-pipelined K loop: 33 tiles of K=32 ----
  for (int ks = 0; ks < NKT; ++ks) {
    __bf16* cur = (ks & 1) ? w1T1 : w1T0;
    __bf16* nxt = (ks & 1) ? w1T0 : w1T1;
    const bool have_next = (ks + 1) < NKT;

    // (1) issue next-tile W1 global loads (bf16, coalesced)
    v8bf s0, s1, s2, s3;
    if (have_next) {
      const __bf16* src = ws + (ks + 1) * (H1 * 32) + tid * 32;
      s0 = *(const v8bf*)(src + 0);
      s1 = *(const v8bf*)(src + 8);
      s2 = *(const v8bf*)(src + 16);
      s3 = *(const v8bf*)(src + 24);
    }
    // (2) issue next A-fragment global loads (fp32)
    v4f r0, r1, r2, r3;
    if (have_next) {
      const bool from_x = (ks + 1) < (NKT - 1);
      const float* p = from_x ? (xrow + (ks + 1) * 32 + 8 * hi)
                              : (task_emb + mytask * TE + 8 * hi);
      r0 = *(const v4f*)(p + 0);
      r1 = *(const v4f*)(p + 4);
      if (from_x) {
        r2 = *(const v4f*)(p + 16);
        r3 = *(const v4f*)(p + 20);
      } else {
#pragma unroll
        for (int j = 0; j < 4; ++j) { r2[j] = 0.0f; r3[j] = 0.0f; }
      }
    }
    // (3) compute: 16 WMMAs with a 4-deep B-fragment queue so each
    //     s_wait_dscnt only needs the oldest pair (LDS latency overlaps WMMA)
    {
      const __bf16* bbase = cur + l16 * W1T_KS + hi * 8;
      v8bf bq[8];
#pragma unroll
      for (int i = 0; i < 4; ++i) {
        const __bf16* bp = bbase + i * (16 * W1T_KS);
        bq[2 * i]     = *(const v8bf*)bp;
        bq[2 * i + 1] = *(const v8bf*)(bp + 16);
      }
#pragma unroll
      for (int nt = 0; nt < 16; ++nt) {
        v16bf bcur = cat8(bq[2 * (nt & 3)], bq[2 * (nt & 3) + 1]);
        if (nt + 4 < 16) {
          const __bf16* bp = bbase + (nt + 4) * (16 * W1T_KS);
          bq[2 * (nt & 3)]     = *(const v8bf*)bp;
          bq[2 * (nt & 3) + 1] = *(const v8bf*)(bp + 16);
        }
        acc1[nt] = __builtin_amdgcn_wmma_f32_16x16x32_bf16(
            false, a_cur, false, bcur, (short)0, acc1[nt], false, false);
      }
    }
    // (4) convert next A fragment (global latency hidden by WMMAs)
    if (have_next) {
      cvt_f4(a_cur, 0, r0);
      cvt_f4(a_cur, 4, r1);
      cvt_f4(a_cur, 8, r2);
      cvt_f4(a_cur, 12, r3);
      // (5) store staged tile to the other LDS buffer
      __bf16* d = nxt + tid * W1T_KS;
      *(v8bf*)(d + 0)  = s0;
      *(v8bf*)(d + 8)  = s1;
      *(v8bf*)(d + 16) = s2;
      *(v8bf*)(d + 24) = s3;
    }
    __syncthreads();
  }

  // ---- bias + ReLU, write h1 (bf16) to per-wave LDS (A-fragment layout) ----
#pragma unroll
  for (int nt = 0; nt < 16; ++nt) {
    float bias = b1[nt * 16 + l16];
#pragma unroll
    for (int v = 0; v < 8; ++v) {
      float h = fmaxf(acc1[nt][v] + bias, 0.0f);
      h1w[(v + 8 * hi) * H1_RS + nt * 16 + l16] = (__bf16)h;
    }
  }

  // ---------------- layer 2: h1[256] @ W2 -> h2[64] ----------------
  v8f acc2[4];
#pragma unroll
  for (int i = 0; i < 4; ++i)
#pragma unroll
    for (int j = 0; j < 8; ++j) acc2[i][j] = 0.0f;

#pragma unroll
  for (int kb = 0; kb < H1; kb += 32) {
    const __bf16* ap = h1w + l16 * H1_RS + kb + 8 * hi;
    v16bf a2 = cat8(*(const v8bf*)ap, *(const v8bf*)(ap + 16));
#pragma unroll
    for (int nt = 0; nt < 4; ++nt) {
      const __bf16* bp = w2T + (nt * 16 + l16) * W2T_KS + kb + hi * 8;
      v16bf bf = cat8(*(const v8bf*)bp, *(const v8bf*)(bp + 16));
      acc2[nt] = __builtin_amdgcn_wmma_f32_16x16x32_bf16(
          false, a2, false, bf, (short)0, acc2[nt], false, false);
    }
  }

  // ---- head directly from registers: bias+ReLU, dot with head_W[task] ----
  float bias2[4];
#pragma unroll
  for (int nt = 0; nt < 4; ++nt) bias2[nt] = b2[nt * 16 + l16];

#pragma unroll
  for (int v = 0; v < 8; ++v) {
    const int row = row_wave + v + 8 * hi;
    const int t = __shfl(mytask, v + 8 * hi, 32);  // wave already holds all 16 tids
    const float* hw = head_W + t * H2 + l16;
    float p = 0.0f;
#pragma unroll
    for (int nt = 0; nt < 4; ++nt) {
      float h = fmaxf(acc2[nt][v] + bias2[nt], 0.0f);
      p += h * hw[nt * 16];
    }
    // reduce over the 16 lanes of this half-wave
    p += __shfl_xor(p, 8, 32);
    p += __shfl_xor(p, 4, 32);
    p += __shfl_xor(p, 2, 32);
    p += __shfl_xor(p, 1, 32);
    if (l16 == 0) out[row] = p + head_b[t];
  }
}

extern "C" void kernel_launch(void* const* d_in, const int* in_sizes, int n_in,
                              void* d_out, int out_size, void* d_ws, size_t ws_size,
                              hipStream_t stream) {
  (void)in_sizes; (void)n_in; (void)out_size; (void)ws_size;
  const float* x        = (const float*)d_in[0];
  const int*   task_ids = (const int*)d_in[1];
  const float* task_emb = (const float*)d_in[2];
  const float* W1       = (const float*)d_in[3];
  const float* b1       = (const float*)d_in[4];
  const float* W2       = (const float*)d_in[5];
  const float* b2       = (const float*)d_in[6];
  const float* head_W   = (const float*)d_in[7];
  const float* head_b   = (const float*)d_in[8];
  float* out = (float*)d_out;
  __bf16* ws = (__bf16*)d_ws;

  // 1) convert W1/W2 to bf16, tiled col-major, zero-padded tail (573 KB in d_ws)
  SharedTaskMLP_wconv_kernel<<<dim3(WS_TOTAL / 256), dim3(256), 0, stream>>>(
      W1, W2, ws);

  // 2) fused MLP
  hipFuncSetAttribute((const void*)SharedTaskMLP_70162585747915_kernel,
                      hipFuncAttributeMaxDynamicSharedMemorySize, SMEM_BYTES);
  SharedTaskMLP_70162585747915_kernel<<<dim3(B_TOT / M_WG), dim3(BLOCK_T),
                                        SMEM_BYTES, stream>>>(
      x, task_ids, task_emb, b1, b2, head_W, head_b, ws, out);
}
